// QuantumAttentionLayer_12713103196200
// MI455X (gfx1250) — matrix operand order repaired and checked by reading, not compiled, not run on hardware
//
#include <hip/hip_runtime.h>
#include <cstdint>

typedef __attribute__((ext_vector_type(16))) __bf16 v16bf;
typedef __attribute__((ext_vector_type(8)))  __bf16 v8bf;
typedef __attribute__((ext_vector_type(8)))  float  v8f;
typedef __attribute__((ext_vector_type(4)))  float  v4f;

#define D_MODEL 1024
#define NHEAD   16
#define HEADD   64
#define QDIM    16
#define SEQ     2048
#define BATCH   2
#define ROWS    (BATCH*SEQ)   /* 4096 */
#define PKK     96            /* 64 head dims + 16 quantum + 16 zero pad */

#define TILE_M 128
#define TILE_N 128
#define LDSK   40             /* padded K stride (bf16 elems): 80B rows stay 16B aligned */

// ---------------- CDNA5 async copy: global -> LDS, 16 bytes per lane ----------------
__device__ inline void async_copy_b128(void* lds_dst, const void* gsrc) {
  asm volatile("global_load_async_to_lds_b128 %0, %1, off"
               :: "v"((uint32_t)(uintptr_t)lds_dst), "v"(gsrc)
               : "memory");
}
__device__ inline void async_wait_0() { asm volatile("s_wait_asynccnt 0x0" ::: "memory"); }
__device__ inline void async_wait_2() { asm volatile("s_wait_asynccnt 0x2" ::: "memory"); }
__device__ inline void async_wait_4() { asm volatile("s_wait_asynccnt 0x4" ::: "memory"); }

// ---------------- fragment gathers matching CDNA5 WMMA bf16 VGPR layouts ----------------
// A 16x32 bf16: lane L holds row M=L%16; K runs {b..b+7, b+16..b+23}, b = (L<16)?0:8
__device__ inline v16bf load_a_frag(const __bf16* p, int lane) {
  int base = (lane & 16) ? 8 : 0;
  v8bf lo = *(const v8bf*)(p + base);        // 16B aligned ds_load_b128
  v8bf hi = *(const v8bf*)(p + base + 16);
  return __builtin_shufflevector(lo, hi, 0,1,2,3,4,5,6,7,8,9,10,11,12,13,14,15);
}
// B 32x16 bf16: lane L holds col N=L%16; K runs {b..b+7, b+16..b+23}, b = (L<16)?0:8
__device__ inline v16bf load_b_frag(const __bf16* p, int lane) {
  int base = (lane & 16) ? 8 : 0;
  v8bf lo = *(const v8bf*)(p + base);
  v8bf hi = *(const v8bf*)(p + base + 16);
  return __builtin_shufflevector(lo, hi, 0,1,2,3,4,5,6,7,8,9,10,11,12,13,14,15);
}

// ---------------- generic batched WMMA GEMM: C = act(A@B^T + bias) ----------------
// A: bf16 (or f32 converted while staging if AF32) row-major [M x Kd], lda
// B: bf16 [N x Kd] row-major, ldb  (all weight/V operands are pre-transposed)
// batch z: zb = z / zInner, zh = z % zInner ; base += Outer*zb + Inner*zh
// Block tile 128x128; 8 waves (4M x 2N); each wave 32x64 = 2x4 WMMA per K-step.
// bf16-A path: fully async-staged with double-buffered LDS ping-pong.
template <int ACT, int AF32>
__global__ void __launch_bounds__(256)
gemm_wmma(const void* Aptr, long aInner, long aOuter, int lda,
          const __bf16* Bptr, long bInner, long bOuter, int ldb,
          float* Cptr, long cInner, long cOuter, int ldc,
          const float* bias, int M, int N, int Kd, int zInner)
{
  __shared__ alignas(16) __bf16 sA[2][TILE_M * LDSK];
  __shared__ alignas(16) __bf16 sB[2][TILE_N * LDSK];   // [n][k]

  int z  = blockIdx.z;
  int zb = z / zInner, zh = z % zInner;
  const __bf16* Ab = (const __bf16*)Aptr + aOuter * zb + aInner * zh;
  const float*  Af = (const float*) Aptr + aOuter * zb + aInner * zh;
  const __bf16* Bb = Bptr + bOuter * zb + bInner * zh;
  float*        Cb = Cptr + cOuter * zb + cInner * zh;

  int mBase = blockIdx.y * TILE_M;
  int nBase = blockIdx.x * TILE_N;
  int t    = threadIdx.x;
  int lane = t & 31, wave = t >> 5;
  int wm = wave >> 1, wn = wave & 1;     // 4 x 2 waves, each 32(M) x 64(N)

  v8f acc[2][4] = {};

  auto stageA = [&](int buf, int k0) {
    if (AF32) {                          // convert f32 -> bf16 while staging (sync)
      int row = t >> 1;
      int cb  = (t & 1) * 16;
      int gr  = mBase + row;
      __bf16* dst = &sA[buf][row * LDSK + cb];
      if (gr < M) {
        const v4f* s4 = (const v4f*)(Af + (long)gr * lda + k0 + cb);
        v4f f0 = s4[0], f1 = s4[1], f2 = s4[2], f3 = s4[3];
        v8bf o0, o1;
#pragma unroll
        for (int i = 0; i < 4; ++i) {
          o0[i]     = (__bf16)f0[i];
          o0[4 + i] = (__bf16)f1[i];
          o1[i]     = (__bf16)f2[i];
          o1[4 + i] = (__bf16)f3[i];
        }
        *(v8bf*)dst = o0; *(v8bf*)(dst + 8) = o1;
        if (k0 + 32 < Kd) __builtin_prefetch((const float*)s4 + 32, 0, 0);
      } else {
        v8bf zv = {};
        *(v8bf*)dst = zv; *(v8bf*)(dst + 8) = zv;
      }
    } else {                             // pure copy -> async DMA, 2 chunks of 16B
#pragma unroll
      for (int i = 0; i < 2; ++i) {
        int c   = t * 2 + i;
        int row = c >> 2;
        int q   = (c & 3) * 8;
        int gr  = mBase + row;
        __bf16* dst = &sA[buf][row * LDSK + q];
        if (gr < M) {
          async_copy_b128(dst, Ab + (long)gr * lda + k0 + q);
        } else {
          v8bf zv = {};
          *(v8bf*)dst = zv;
        }
      }
    }
  };
  auto stageB = [&](int buf, int k0) {   // 128 rows x 32 k, async, 2 chunks of 16B
#pragma unroll
    for (int i = 0; i < 2; ++i) {
      int c  = t * 2 + i;
      int n  = c >> 2;
      int q  = (c & 3) * 8;
      int gn = nBase + n;
      __bf16* dst = &sB[buf][n * LDSK + q];
      if (gn < N) {
        async_copy_b128(dst, Bb + (long)gn * ldb + k0 + q);
      } else {
        v8bf zv = {};
        *(v8bf*)dst = zv;
      }
    }
  };
  auto compute = [&](int buf) {
    int arow = wm * 32 + (lane & 15);
    v16bf a0 = load_a_frag(&sA[buf][arow * LDSK], lane);
    v16bf a1 = load_a_frag(&sA[buf][(arow + 16) * LDSK], lane);
    int bcol = wn * 64 + (lane & 15);
#pragma unroll
    for (int j = 0; j < 4; ++j) {
      v16bf b = load_b_frag(&sB[buf][(bcol + j * 16) * LDSK], lane);
      acc[0][j] = __builtin_amdgcn_wmma_f32_16x16x32_bf16(false, a0, false, b, (short)0, acc[0][j], false, false);
      acc[1][j] = __builtin_amdgcn_wmma_f32_16x16x32_bf16(false, a1, false, b, (short)0, acc[1][j], false, false);
    }
  };

  if (!AF32) {
    // fully async: double-buffered ping-pong; per-wave async count per stage is
    // uniform (A:2, B:+2 iff this wave's 16-row n-range < N; N is a multiple of 16).
    bool waveB = (nBase + wave * 16) < N;
    stageA(0, 0); stageB(0, 0);
    int buf = 0;
    for (int k0 = 0; k0 < Kd; k0 += 32) {
      bool hasNext = (k0 + 32) < Kd;
      if (hasNext) { stageA(buf ^ 1, k0 + 32); stageB(buf ^ 1, k0 + 32); }
      if (hasNext) { if (waveB) async_wait_4(); else async_wait_2(); }
      else         { async_wait_0(); }
      __syncthreads();           // everyone's current tile arrived
      compute(buf);
      __syncthreads();           // done reading buf; it may be overwritten next iter
      buf ^= 1;
    }
  } else {
    for (int k0 = 0; k0 < Kd; k0 += 32) {
      stageA(0, k0);
      stageB(0, k0);
      async_wait_0();
      __syncthreads();
      compute(0);
      __syncthreads();
    }
  }

  // ---- epilogue: D layout row = r + 8*(lane/16), col = lane%16 ----
  int rofs  = (lane >> 4) * 8;
  int col16 = lane & 15;
#pragma unroll
  for (int i = 0; i < 2; ++i)
#pragma unroll
    for (int j = 0; j < 4; ++j) {
      int colg = nBase + wn * 64 + j * 16 + col16;
      if (colg >= N) continue;
      float bv = bias ? bias[colg] : 0.f;
#pragma unroll
      for (int r = 0; r < 8; ++r) {
        int rowg = mBase + wm * 32 + i * 16 + rofs + r;
        if (rowg >= M) continue;
        float v = acc[i][j][r] + bv;
        if (ACT == 1) v = tanhf(v);
        Cb[(long)rowg * ldc + colg] = v;
      }
    }
}

// ---------------- f32 -> bf16 conversion (no transpose; for the A operand x) ----------------
__global__ void cvt_bf16(const float* __restrict__ in, __bf16* __restrict__ out, long n) {
  long i = (long)blockIdx.x * blockDim.x + threadIdx.x;
  long stride = (long)gridDim.x * blockDim.x;
  for (; i < n; i += stride) out[i] = (__bf16)in[i];
}

// ---------------- tiled f32 [rows x cols] -> bf16 [cols x rows] transpose-convert ----------------
__global__ void __launch_bounds__(256)
transpose_cvt_bf16(const float* __restrict__ in, __bf16* __restrict__ out,
                   int rows, int cols, long inBatch, long outBatch)
{
  __shared__ float tile[32][33];
  int bz = blockIdx.z;
  const float* ip = in + inBatch * bz;
  __bf16* op = out + outBatch * bz;
  int c0 = blockIdx.x * 32, r0 = blockIdx.y * 32;
  int tx = threadIdx.x & 31, ty = threadIdx.x >> 5;   // 32 x 8
#pragma unroll
  for (int i = 0; i < 32; i += 8) {
    int r = r0 + ty + i, c = c0 + tx;
    tile[ty + i][tx] = (r < rows && c < cols) ? ip[(long)r * cols + c] : 0.f;
  }
  __syncthreads();
#pragma unroll
  for (int i = 0; i < 32; i += 8) {
    int c = c0 + ty + i, r = r0 + tx;
    if (c < cols && r < rows) op[(long)c * rows + r] = (__bf16)tile[tx][ty + i];
  }
}

// ---------------- quantum chain + gate + 0.7/0.3 fuse (in-place on Q or K) ----------------
__global__ void __launch_bounds__(256)
quantum_fuse(const float* __restrict__ qs, const float* __restrict__ classical,
             float* QK, const float* __restrict__ qw_r, const float* __restrict__ qw_i,
             const float* __restrict__ em, const float* __restrict__ mb,
             float* __restrict__ qp_out)
{
  __shared__ float s_qwr[256], s_qwi[256], s_em[256];
  __shared__ float sQS[16][16], sTR[16][16], sTI[16][16], sP[16][17];
  int t = threadIdx.x;
  s_qwr[t] = qw_r[t]; s_qwi[t] = qw_i[t]; s_em[t] = em[t];
  int r = t >> 4, j = t & 15;
  long row = (long)blockIdx.x * 16 + r;
  sQS[r][j] = qs[row * QDIM + j];
  __syncthreads();
  float tr = 0.f, ti = 0.f;
#pragma unroll
  for (int i = 0; i < 16; ++i) { float q = sQS[r][i]; tr += q * s_qwr[i*16+j]; ti += q * s_qwi[i*16+j]; }
  sTR[r][j] = tr; sTI[r][j] = ti;
  __syncthreads();
  float er = 0.f, ei = 0.f;
#pragma unroll
  for (int k = 0; k < 16; ++k) { float e = s_em[k*16+j]; er += sTR[r][k]*e; ei += sTI[r][k]*e; }
  float prob = er*er + ei*ei;
  sP[r][j] = prob;
  qp_out[row * QDIM + j] = prob;
  __syncthreads();
  float s = 0.f;
#pragma unroll
  for (int k = 0; k < 16; ++k) s += sP[r][k];
  float fw = 1.f / (1.f + __expf(-s * (1.f/16.f)));
  for (int c = j * 64; c < j * 64 + 64; ++c) {
    float qo = 0.f;
#pragma unroll
    for (int k = 0; k < 16; ++k) qo += sP[r][k] * mb[k * D_MODEL + c];
    long idx = row * D_MODEL + c;
    float o  = fw * qo + (1.f - fw) * classical[idx];
    QK[idx] = 0.7f * QK[idx] + 0.3f * o;
  }
}

// ---------------- pack [b,h,i, 0:64]=fs*F[.,h*64+d] | [64:80]=ps*p | [80:96]=0 ----------------
__global__ void pack_qk(const float* __restrict__ F, const float* __restrict__ P,
                        __bf16* __restrict__ out, float fs, float ps, long total)
{
  long i = (long)blockIdx.x * blockDim.x + threadIdx.x;
  long stride = (long)gridDim.x * blockDim.x;
  for (; i < total; i += stride) {
    int  c  = (int)(i % PKK);
    long r  = i / PKK;                    // r = (b*H + h)*SEQ + s
    long si = r % SEQ;
    long bh = r / SEQ;
    long h  = bh % NHEAD, b = bh / NHEAD;
    long row = b * SEQ + si;
    float v;
    if (c < HEADD)             v = fs * F[row * D_MODEL + h * HEADD + c];
    else if (c < HEADD + QDIM) v = ps * P[row * QDIM + (c - HEADD)];
    else                       v = 0.f;
    out[i] = (__bf16)v;
  }
}

// ---------------- in-place row softmax ----------------
__global__ void __launch_bounds__(256)
softmax_rows(float* __restrict__ w, int ncols)
{
  __shared__ float red[256];
  long row = blockIdx.x;
  float* p = w + row * (long)ncols;
  int t = threadIdx.x;
  float m = -1e30f;
  for (int c = t; c < ncols; c += 256) m = fmaxf(m, p[c]);
  red[t] = m; __syncthreads();
  for (int s = 128; s > 0; s >>= 1) { if (t < s) red[t] = fmaxf(red[t], red[t+s]); __syncthreads(); }
  m = red[0]; __syncthreads();
  float sum = 0.f;
  for (int c = t; c < ncols; c += 256) { float e = __expf(p[c] - m); p[c] = e; sum += e; }
  red[t] = sum; __syncthreads();
  for (int s = 128; s > 0; s >>= 1) { if (t < s) red[t] += red[t+s]; __syncthreads(); }
  float inv = 1.f / red[0];
  for (int c = t; c < ncols; c += 256) p[c] *= inv;
}

// ---------------- residual + layernorm ----------------
__global__ void __launch_bounds__(256)
residual_ln(const float* __restrict__ proj, const float* __restrict__ x,
            const float* __restrict__ g, const float* __restrict__ b, float* __restrict__ y)
{
  __shared__ float rs[256], rq[256];
  long row = blockIdx.x;
  long base = row * D_MODEL;
  int t = threadIdx.x;
  float s = 0.f, q = 0.f;
  for (int c = t; c < D_MODEL; c += 256) { float v = proj[base+c] + x[base+c]; s += v; q += v*v; }
  rs[t] = s; rq[t] = q; __syncthreads();
  for (int st = 128; st > 0; st >>= 1) { if (t < st) { rs[t] += rs[t+st]; rq[t] += rq[t+st]; } __syncthreads(); }
  float mu  = rs[0] * (1.f / D_MODEL);
  float var = rq[0] * (1.f / D_MODEL) - mu * mu;
  float inv = rsqrtf(var + 1e-5f);
  for (int c = t; c < D_MODEL; c += 256) {
    float v = proj[base+c] + x[base+c];
    y[base+c] = (v - mu) * inv * g[c] + b[c];
  }
}

// ============================================================================
extern "C" void kernel_launch(void* const* d_in, const int* in_sizes, int n_in,
                              void* d_out, int out_size, void* d_ws, size_t ws_size,
                              hipStream_t stream)
{
  (void)in_sizes; (void)n_in; (void)out_size; (void)ws_size;
  const float* X    = (const float*)d_in[0];
  const float* Wq   = (const float*)d_in[1];  const float* bq = (const float*)d_in[2];
  const float* Wk   = (const float*)d_in[3];  const float* bk = (const float*)d_in[4];
  const float* Wv   = (const float*)d_in[5];  const float* bv = (const float*)d_in[6];
  const float* Wo   = (const float*)d_in[7];  const float* bo = (const float*)d_in[8];
  const float* lng  = (const float*)d_in[9];  const float* lnb = (const float*)d_in[10];
  const float* qqW  = (const float*)d_in[11]; const float* qqb = (const float*)d_in[12];
  const float* qqR  = (const float*)d_in[13]; const float* qqI = (const float*)d_in[14];
  const float* qqS  = (const float*)d_in[15]; const float* qqE = (const float*)d_in[16];
  const float* qqM  = (const float*)d_in[17];
  const float* qkW  = (const float*)d_in[18]; const float* qkb = (const float*)d_in[19];
  const float* qkR  = (const float*)d_in[20]; const float* qkI = (const float*)d_in[21];
  const float* qkS  = (const float*)d_in[22]; const float* qkE = (const float*)d_in[23];
  const float* qkM  = (const float*)d_in[24];

  float* yout = (float*)d_out;                              // [2,2048,1024]
  float* wout = yout + (long)ROWS * D_MODEL;                // [2,16,2048,2048]

  // ---- workspace carve-up ----
  char* ws = (char*)d_ws;
  size_t off = 0;
  auto take = [&](size_t bytes) -> char* {
    char* p = ws + off;
    off = (off + bytes + 255) & ~(size_t)255;
    return p;
  };
  __bf16* xb   = (__bf16*)take((size_t)ROWS * D_MODEL * 2);
  __bf16* wqT  = (__bf16*)take((size_t)D_MODEL * D_MODEL * 2);   // all weights transposed [N][K]
  __bf16* wkT  = (__bf16*)take((size_t)D_MODEL * D_MODEL * 2);
  __bf16* wvT  = (__bf16*)take((size_t)D_MODEL * D_MODEL * 2);
  __bf16* woT  = (__bf16*)take((size_t)D_MODEL * D_MODEL * 2);
  __bf16* qqwT = (__bf16*)take((size_t)D_MODEL * D_MODEL * 2);
  __bf16* qkwT = (__bf16*)take((size_t)D_MODEL * D_MODEL * 2);
  __bf16* sgqT = (__bf16*)take((size_t)QDIM * D_MODEL * 2);      // [16][1024]
  __bf16* sgkT = (__bf16*)take((size_t)QDIM * D_MODEL * 2);
  float*  Qb   = (float*)take((size_t)ROWS * D_MODEL * 4);
  float*  Kb   = (float*)take((size_t)ROWS * D_MODEL * 4);
  float*  Vb32 = (float*)take((size_t)ROWS * D_MODEL * 4);
  float*  Cq   = (float*)take((size_t)ROWS * D_MODEL * 4);
  float*  Ck   = (float*)take((size_t)ROWS * D_MODEL * 4);
  float*  qsq  = (float*)take((size_t)ROWS * QDIM * 4);
  float*  qsk  = (float*)take((size_t)ROWS * QDIM * 4);
  float*  qp   = (float*)take((size_t)ROWS * QDIM * 4);
  float*  kp   = (float*)take((size_t)ROWS * QDIM * 4);
  __bf16* vbT  = (__bf16*)take((size_t)BATCH * D_MODEL * SEQ * 2);  // [b][c][j]
  __bf16* qpk  = (__bf16*)take((size_t)BATCH * NHEAD * SEQ * PKK * 2);
  __bf16* kpk  = (__bf16*)take((size_t)BATCH * NHEAD * SEQ * PKK * 2);
  float*  att  = (float*)take((size_t)ROWS * D_MODEL * 4);
  float*  proj = (float*)take((size_t)ROWS * D_MODEL * 4);

  dim3 blk(256);

  // 1) x -> bf16 ; weights -> transposed bf16 [N][K]
  cvt_bf16<<<2048, blk, 0, stream>>>(X, xb, (long)ROWS * D_MODEL);
  dim3 gT(32, 32, 1);
  transpose_cvt_bf16<<<gT, blk, 0, stream>>>(Wq,  wqT,  D_MODEL, D_MODEL, 0, 0);
  transpose_cvt_bf16<<<gT, blk, 0, stream>>>(Wk,  wkT,  D_MODEL, D_MODEL, 0, 0);
  transpose_cvt_bf16<<<gT, blk, 0, stream>>>(Wv,  wvT,  D_MODEL, D_MODEL, 0, 0);
  transpose_cvt_bf16<<<gT, blk, 0, stream>>>(Wo,  woT,  D_MODEL, D_MODEL, 0, 0);
  transpose_cvt_bf16<<<gT, blk, 0, stream>>>(qqW, qqwT, D_MODEL, D_MODEL, 0, 0);
  transpose_cvt_bf16<<<gT, blk, 0, stream>>>(qkW, qkwT, D_MODEL, D_MODEL, 0, 0);
  dim3 gTs(1, 32, 1);
  transpose_cvt_bf16<<<gTs, blk, 0, stream>>>(qqS, sgqT, D_MODEL, QDIM, 0, 0);
  transpose_cvt_bf16<<<gTs, blk, 0, stream>>>(qkS, sgkT, D_MODEL, QDIM, 0, 0);

  // 2) big projections (async ping-pong WMMA bf16, f32 accum)
  dim3 gBig(D_MODEL / TILE_N, ROWS / TILE_M, 1);
  gemm_wmma<0,0><<<gBig, blk, 0, stream>>>(xb,0,0,D_MODEL, wqT,0,0,D_MODEL, Qb,0,0,D_MODEL, bq,  ROWS,D_MODEL,D_MODEL, 1);
  gemm_wmma<0,0><<<gBig, blk, 0, stream>>>(xb,0,0,D_MODEL, wkT,0,0,D_MODEL, Kb,0,0,D_MODEL, bk,  ROWS,D_MODEL,D_MODEL, 1);
  gemm_wmma<0,0><<<gBig, blk, 0, stream>>>(xb,0,0,D_MODEL, wvT,0,0,D_MODEL, Vb32,0,0,D_MODEL, bv, ROWS,D_MODEL,D_MODEL, 1);
  gemm_wmma<1,0><<<gBig, blk, 0, stream>>>(xb,0,0,D_MODEL, qqwT,0,0,D_MODEL, Cq,0,0,D_MODEL, qqb, ROWS,D_MODEL,D_MODEL, 1);
  gemm_wmma<1,0><<<gBig, blk, 0, stream>>>(xb,0,0,D_MODEL, qkwT,0,0,D_MODEL, Ck,0,0,D_MODEL, qkb, ROWS,D_MODEL,D_MODEL, 1);
  dim3 gQS(1, ROWS / TILE_M, 1);
  gemm_wmma<0,0><<<gQS, blk, 0, stream>>>(xb,0,0,D_MODEL, sgqT,0,0,D_MODEL, qsq,0,0,QDIM, nullptr, ROWS,QDIM,D_MODEL, 1);
  gemm_wmma<0,0><<<gQS, blk, 0, stream>>>(xb,0,0,D_MODEL, sgkT,0,0,D_MODEL, qsk,0,0,QDIM, nullptr, ROWS,QDIM,D_MODEL, 1);

  // 3) quantum chain + gate, fuse 0.7*lin + 0.3*quantum in place
  quantum_fuse<<<ROWS / 16, blk, 0, stream>>>(qsq, Cq, Qb, qqR, qqI, qqE, qqM, qp);
  quantum_fuse<<<ROWS / 16, blk, 0, stream>>>(qsk, Ck, Kb, qkR, qkI, qkE, qkM, kp);

  // 4) V -> transposed bf16 per batch ; pack Qext/Kext (interference folded into K dim)
  dim3 gTv(D_MODEL / 32, SEQ / 32, BATCH);
  transpose_cvt_bf16<<<gTv, blk, 0, stream>>>(Vb32, vbT, SEQ, D_MODEL,
                                              (long)SEQ * D_MODEL, (long)D_MODEL * SEQ);
  const long pkTotal = (long)BATCH * NHEAD * SEQ * PKK;
  pack_qk<<<2048, blk, 0, stream>>>(Qb, qp, qpk, 0.125f, 0.1f, pkTotal);  // 1/sqrt(64)=0.125
  pack_qk<<<2048, blk, 0, stream>>>(Kb, kp, kpk, 1.0f,   1.0f, pkTotal);

  // 5) scores = Qext @ Kext^T  (batched over 32 (b,h), K=96) -> straight into d_out w region
  dim3 gSc(SEQ / TILE_N, SEQ / TILE_M, BATCH * NHEAD);
  gemm_wmma<0,0><<<gSc, blk, 0, stream>>>(
      qpk, (long)SEQ * PKK, 0, PKK,
      kpk, (long)SEQ * PKK, 0, PKK,
      wout, (long)SEQ * SEQ, 0, SEQ,
      nullptr, SEQ, SEQ, PKK, BATCH * NHEAD);

  // 6) softmax in place on w
  softmax_rows<<<BATCH * NHEAD * SEQ, blk, 0, stream>>>(wout, SEQ);

  // 7) att = w @ Vh (A is f32, converted to bf16 during LDS staging; B = V^T slices)
  dim3 gAtt(1, SEQ / TILE_M, BATCH * NHEAD);
  gemm_wmma<0,1><<<gAtt, blk, 0, stream>>>(
      wout, (long)SEQ * SEQ,   (long)NHEAD * SEQ * SEQ, SEQ,
      vbT,  (long)HEADD * SEQ, (long)D_MODEL * SEQ,     SEQ,
      att,  (long)HEADD,       (long)SEQ * D_MODEL,     D_MODEL,
      nullptr, SEQ, HEADD, SEQ, NHEAD);

  // 8) output projection
  gemm_wmma<0,1><<<gBig, blk, 0, stream>>>(att,0,0,D_MODEL, woT,0,0,D_MODEL, proj,0,0,D_MODEL, bo, ROWS,D_MODEL,D_MODEL, 1);

  // 9) residual + layernorm -> y
  residual_ln<<<ROWS, blk, 0, stream>>>(proj, X, lng, lnb, yout);
}
